// MultiHeadAttention_62603443306502
// MI455X (gfx1250) — matrix operand order repaired and checked by reading, not compiled, not run on hardware
//
#include <hip/hip_runtime.h>
#include <hip/hip_bf16.h>
#include <stdint.h>

// MHA: B=4, S=2048, D=1024, H=16, Dh=64
#define D_MODEL   1024
#define NUM_HEADS 16
#define HEAD_DIM  64
#define SEQ       2048
#define BATCH     4
#define ROWS      (BATCH * SEQ)   // 8192 tokens

typedef __attribute__((ext_vector_type(16))) __bf16 v16bf;
typedef __attribute__((ext_vector_type(8)))  float  v8f;
typedef uint32_t u32x4 __attribute__((ext_vector_type(4)));
typedef uint32_t u32x8 __attribute__((ext_vector_type(8)));

union FragB {
  v16bf v;
  uint4 u[2];
};

__device__ __forceinline__ uint16_t f2bf(float f) {
  union { float f; uint32_t u; } c;
  c.f = f;
  uint32_t u = c.u;
  return (uint16_t)((u + 0x7FFFu + ((u >> 16) & 1u)) >> 16);  // RNE
}

// Low 32 bits of a generic pointer to __shared__ = LDS byte offset
__device__ __forceinline__ uint32_t lds_off(const void* p) {
  return (uint32_t)(uintptr_t)p;
}

__device__ __forceinline__ uint32_t rfl(uint32_t v) {
  return (uint32_t)__builtin_amdgcn_readfirstlane((int)v);
}

// ---------------------------------------------------------------------------
// TDM: issue a 2D tensor-tile DMA  global -> LDS  (bf16 elements).
// D# packing per CDNA5 ISA ch.8:
//  group0: [1:0]=count=1 | [63:32]=lds_addr | [120:64]=global_addr | type=2
//  group1: data_size=1(2B) @ [17:16]; tensor_dim0 @ [79:48]; tensor_dim1 @
//          [111:80]; tile_dim0 @ [127:112]; tile_dim1 @ [143:128];
//          tensor_dim0_stride @ [207:160]
// Uniform inputs are forced into SGPRs via readfirstlane so the descriptor
// groups build scalar-side. Caller ensures only ONE wave executes this.
// ---------------------------------------------------------------------------
__device__ __forceinline__ void tdm_load_2d(const void* gptr, uint32_t ldsoff,
                                            uint32_t tile_d0, uint32_t tile_d1,
                                            uint32_t tensor_d0,
                                            uint32_t tensor_d1,
                                            uint32_t stride_elems) {
  uint64_t ga = (uint64_t)(uintptr_t)gptr;
  uint32_t ga_lo = rfl((uint32_t)ga);
  uint32_t ga_hi = rfl((uint32_t)(ga >> 32));
  u32x4 g0;
  g0[0] = 0x1u;                                    // count=1
  g0[1] = rfl(ldsoff);                             // lds_addr
  g0[2] = ga_lo;                                   // addr[31:0]
  g0[3] = (ga_hi & 0x1FFFFFFu) | 0x80000000u;      // addr[56:32] | type=2
  u32x8 g1;
  g1[0] = 1u << 16;                                // data_size=2B
  g1[1] = (tensor_d0 & 0xFFFFu) << 16;             // tensor_dim0[15:0]
  g1[2] = (tensor_d0 >> 16) | ((tensor_d1 & 0xFFFFu) << 16);
  g1[3] = (tensor_d1 >> 16) | (tile_d0 << 16);     // tile_dim0
  g1[4] = tile_d1 & 0xFFFFu;                       // tile_dim1 (tile_dim2=0)
  g1[5] = stride_elems;                            // tensor_dim0_stride
  g1[6] = 0u;
  g1[7] = 0u;
  asm volatile("tensor_load_to_lds %0, %1" ::"s"(g0), "s"(g1) : "memory");
}

// ---------------------------------------------------------------------------
// fp32 -> bf16 conversion (grid-stride)
// ---------------------------------------------------------------------------
__global__ __launch_bounds__(256) void cvt_f32_bf16(const float* __restrict__ in,
                                                    uint16_t* __restrict__ out,
                                                    int n) {
  int i = blockIdx.x * blockDim.x + threadIdx.x;
  int stride = gridDim.x * blockDim.x;
  for (; i < n; i += stride) out[i] = f2bf(in[i]);
}

// ---------------------------------------------------------------------------
// GEMM building blocks. Block = 256 thr = 8 waves (2 in M x 4 in N); block
// tile 64(M) x 256(N); per wave 32x64 -> acc[2][4] v8f.
// WMMA operand layouts per ISA 7.12.2:
//   A 16x32: lane r=l&15 -> row; hi=l>>4 -> K chunks {hi*8, 16+hi*8} (2x16B)
//   B 32x16: lane r -> col N; hi -> K chunks {hi*8, 16+hi*8} (2x16B)
//   C/D:     element i -> row i + hi*8, col r
// ---------------------------------------------------------------------------
#define GEMM_ISSUE(Asrc, Wsrc, kk, bufA, bufB)                                 \
  {                                                                            \
    tdm_load_2d((Asrc) + (kk), lds_off(bufA), 32, 64, D_MODEL, ROWS,           \
                D_MODEL);                                                      \
    tdm_load_2d((Wsrc) + (kk), lds_off(bufB), 32, 256, D_MODEL, D_MODEL,       \
                D_MODEL);                                                      \
  }

#define GEMM_COMPUTE(bufA, bufB, acc)                                          \
  {                                                                            \
    FragB a[2];                                                                \
    _Pragma("unroll") for (int s = 0; s < 2; ++s) {                            \
      const uint16_t* row = &(bufA)[(wm * 32 + s * 16 + r) * 32];              \
      a[s].u[0] = *(const uint4*)(row + hi * 8);                               \
      a[s].u[1] = *(const uint4*)(row + 16 + hi * 8);                          \
    }                                                                          \
    _Pragma("unroll") for (int n = 0; n < 4; ++n) {                            \
      FragB bm;                                                                \
      const uint16_t* wrow = &(bufB)[(wn * 64 + n * 16 + r) * 32 + hi * 8];    \
      bm.u[0] = *(const uint4*)(wrow);                                         \
      bm.u[1] = *(const uint4*)(wrow + 16);                                    \
      _Pragma("unroll") for (int s = 0; s < 2; ++s)                            \
          acc[s][n] = __builtin_amdgcn_wmma_f32_16x16x32_bf16(                 \
              false, a[s].v, false, bm.v, (short)0, acc[s][n], false, false);  \
    }                                                                          \
  }

// Double-buffered, TDM-staged K-loop, explicitly unrolled over both buffers
// so every LDS index is a compile-time constant.
#define GEMM_STAGED_BODY(Asrc, Wsrc, acc)                                      \
  {                                                                            \
    if (threadIdx.x < 32) GEMM_ISSUE(Asrc, Wsrc, 0, ldsA[0], ldsB[0])          \
    for (int k = 0; k < D_MODEL; k += 64) {                                    \
      if (threadIdx.x < 32) __builtin_amdgcn_s_wait_tensorcnt(0);              \
      __syncthreads();                                                         \
      if (threadIdx.x < 32)                                                    \
        GEMM_ISSUE(Asrc, Wsrc, k + 32, ldsA[1], ldsB[1])  /* k+32 < 1024 */    \
      GEMM_COMPUTE(ldsA[0], ldsB[0], acc)                                      \
      if (threadIdx.x < 32) __builtin_amdgcn_s_wait_tensorcnt(0);              \
      __syncthreads();                                                         \
      if (threadIdx.x < 32 && (k + 64) < D_MODEL)                              \
        GEMM_ISSUE(Asrc, Wsrc, k + 64, ldsA[0], ldsB[0])                       \
      GEMM_COMPUTE(ldsA[1], ldsB[1], acc)                                      \
    }                                                                          \
  }

// ---------------------------------------------------------------------------
// Fused Q/K/V projection. grid=(ROWS/64, D_MODEL/256, 3); block=256.
// z=0 -> Q bf16 [token,D]; z=1 -> K same; z=2 -> V stored transposed per head:
// Vt[(b*H+h)*Dh + dh][s]  (so flash P*V B-operand is key-contiguous).
// ---------------------------------------------------------------------------
__global__ __launch_bounds__(256) void gemm_qkv(
    const uint16_t* __restrict__ xb,
    const uint16_t* __restrict__ wq, const uint16_t* __restrict__ wk,
    const uint16_t* __restrict__ wv,
    const float* __restrict__ bq, const float* __restrict__ bk,
    const float* __restrict__ bv,
    uint16_t* __restrict__ Qb, uint16_t* __restrict__ Kb,
    uint16_t* __restrict__ Vt) {
  __shared__ __align__(128) uint16_t ldsA[2][64 * 32];    //  8 KB
  __shared__ __align__(128) uint16_t ldsB[2][256 * 32];   // 32 KB

  const int wave = threadIdx.x >> 5;
  const int lane = threadIdx.x & 31;
  const int r = lane & 15;
  const int hi = lane >> 4;
  const int wm = wave >> 2;
  const int wn = wave & 3;
  const int m0 = blockIdx.x * 64 + wm * 32;
  const int n0 = blockIdx.y * 256 + wn * 64;
  const int z = blockIdx.z;

  const uint16_t* W = (z == 0) ? wq : (z == 1) ? wk : wv;
  const float* bias = (z == 0) ? bq : (z == 1) ? bk : bv;
  const uint16_t* Asrc = xb + (size_t)blockIdx.x * 64 * D_MODEL;
  const uint16_t* Wsrc = W + (size_t)blockIdx.y * 256 * D_MODEL;

  v8f acc[2][4];
#pragma unroll
  for (int s = 0; s < 2; ++s)
#pragma unroll
    for (int n = 0; n < 4; ++n) acc[s][n] = (v8f){};

  GEMM_STAGED_BODY(Asrc, Wsrc, acc)

  if (z < 2) {
    uint16_t* out = (z == 0) ? Qb : Kb;
#pragma unroll
    for (int s = 0; s < 2; ++s)
#pragma unroll
      for (int n = 0; n < 4; ++n) {
        const int col = n0 + n * 16 + r;
        const float bb = bias[col];
#pragma unroll
        for (int i = 0; i < 8; ++i) {
          const int row = m0 + s * 16 + i + hi * 8;
          out[(size_t)row * D_MODEL + col] = f2bf(acc[s][n][i] + bb);
        }
      }
  } else {
#pragma unroll
    for (int s = 0; s < 2; ++s)
#pragma unroll
      for (int n = 0; n < 4; ++n) {
        const int col = n0 + n * 16 + r;
        const float bb = bias[col];
        const int h = col >> 6;
        const int dh = col & 63;
#pragma unroll
        for (int i = 0; i < 8; ++i) {
          const int rg = m0 + s * 16 + i + hi * 8;   // global token
          const int bidx = rg >> 11;                 // tiles never straddle batch
          const int sq = rg & (SEQ - 1);
          Vt[(size_t)((bidx * NUM_HEADS + h) * HEAD_DIM + dh) * SEQ + sq] =
              f2bf(acc[s][n][i] + bb);
        }
      }
  }
}

// ---------------------------------------------------------------------------
// Flash attention. Block = 8 waves; all 8 waves share one (b,h). Per 64-key
// step, wave 0 TDM-stages the K tile (64 keys x 64 dh) and V^T tile
// (64 dh x 64 keys) into double-buffered LDS shared by the block.
// Key loop unrolled over both buffers (steps of 128 keys).
// ---------------------------------------------------------------------------
__global__ __launch_bounds__(256) void flash_attn(
    const uint16_t* __restrict__ Qb, const uint16_t* __restrict__ Kb,
    const uint16_t* __restrict__ Vt, uint16_t* __restrict__ Ctx) {
  __shared__ __align__(128) uint16_t ldsK[2][64 * 64];   // 16 KB [key][dh]
  __shared__ __align__(128) uint16_t ldsV[2][64 * 64];   // 16 KB [dh][key]
  __shared__ __align__(16) uint16_t plds[8][16 * 64];    // 16 KB P slabs

  const int wave = threadIdx.x >> 5;
  const int lane = threadIdx.x & 31;
  const int r = lane & 15;
  const int hi = lane >> 4;

  const int gw = blockIdx.x * 8 + wave;
  const int q16 = gw & 127;
  const int bh = gw >> 7;
  const int b = bh >> 4;
  const int h = bh & 15;
  const int q0 = q16 * 16;

  const uint16_t* Ksrc = Kb + (size_t)(b * SEQ) * D_MODEL + h * HEAD_DIM;
  const uint16_t* Vsrc = Vt + (size_t)((b * NUM_HEADS + h) * HEAD_DIM) * SEQ;

  // Q fragments (16 x 64, two K=32 chunks), loaded once from global
  FragB qf[2];
  const uint16_t* qrow = Qb + (size_t)(b * SEQ + q0 + r) * D_MODEL + h * HEAD_DIM;
#pragma unroll
  for (int j = 0; j < 2; ++j) {
    qf[j].u[0] = *(const uint4*)(qrow + j * 32 + hi * 8);
    qf[j].u[1] = *(const uint4*)(qrow + j * 32 + 16 + hi * 8);
  }

  float m_i[8], l_i[8];
  v8f o[4];
#pragma unroll
  for (int i = 0; i < 8; ++i) { m_i[i] = -1e30f; l_i[i] = 0.f; }
#pragma unroll
  for (int n = 0; n < 4; ++n) o[n] = (v8f){};

  uint16_t* pl = plds[wave];
  const float scale = 0.125f;  // 1/sqrt(64)

#define FA_ISSUE(key, bufK, bufV)                                              \
  {                                                                            \
    tdm_load_2d(Ksrc + (size_t)(key) * D_MODEL, lds_off(bufK), 64, 64,         \
                D_MODEL, SEQ, D_MODEL);                                        \
    tdm_load_2d(Vsrc + (key), lds_off(bufV), 64, 64, SEQ, HEAD_DIM, SEQ);      \
  }

#define FA_STEP(bufK, bufV)                                                    \
  {                                                                            \
    v8f sacc[4];                                                               \
    _Pragma("unroll") for (int n = 0; n < 4; ++n) {                            \
      v8f c = (v8f){};                                                         \
      _Pragma("unroll") for (int j = 0; j < 2; ++j) {                          \
        FragB kf;                                                              \
        const uint16_t* krow = &(bufK)[(n * 16 + r) * 64 + j * 32 + hi * 8];   \
        kf.u[0] = *(const uint4*)(krow);                                       \
        kf.u[1] = *(const uint4*)(krow + 16);                                  \
        c = __builtin_amdgcn_wmma_f32_16x16x32_bf16(false, qf[j].v, false,     \
                                                    kf.v, (short)0, c, false,  \
                                                    false);                    \
      }                                                                        \
      sacc[n] = c;                                                             \
    }                                                                          \
    _Pragma("unroll") for (int n = 0; n < 4; ++n)                              \
        _Pragma("unroll") for (int i = 0; i < 8; ++i) sacc[n][i] *= scale;     \
    _Pragma("unroll") for (int i = 0; i < 8; ++i) {                            \
      float mx = fmaxf(fmaxf(sacc[0][i], sacc[1][i]),                          \
                       fmaxf(sacc[2][i], sacc[3][i]));                         \
      mx = fmaxf(mx, __shfl_xor(mx, 1, 32));                                   \
      mx = fmaxf(mx, __shfl_xor(mx, 2, 32));                                   \
      mx = fmaxf(mx, __shfl_xor(mx, 4, 32));                                   \
      mx = fmaxf(mx, __shfl_xor(mx, 8, 32));                                   \
      const float mnew = fmaxf(m_i[i], mx);                                    \
      const float corr = __expf(m_i[i] - mnew);                                \
      float rs = 0.f;                                                          \
      _Pragma("unroll") for (int n = 0; n < 4; ++n) {                          \
        const float pe = __expf(sacc[n][i] - mnew);                            \
        sacc[n][i] = pe;                                                       \
        rs += pe;                                                              \
      }                                                                        \
      rs += __shfl_xor(rs, 1, 32);                                             \
      rs += __shfl_xor(rs, 2, 32);                                             \
      rs += __shfl_xor(rs, 4, 32);                                             \
      rs += __shfl_xor(rs, 8, 32);                                             \
      l_i[i] = l_i[i] * corr + rs;                                             \
      m_i[i] = mnew;                                                           \
      _Pragma("unroll") for (int n = 0; n < 4; ++n) o[n][i] *= corr;           \
    }                                                                          \
    _Pragma("unroll") for (int n = 0; n < 4; ++n)                              \
        _Pragma("unroll") for (int i = 0; i < 8; ++i)                          \
            pl[(i + hi * 8) * 64 + n * 16 + r] = f2bf(sacc[n][i]);             \
    asm volatile("s_wait_dscnt 0x0" ::: "memory");                             \
    FragB pf[2];                                                               \
    _Pragma("unroll") for (int j = 0; j < 2; ++j) {                            \
      pf[j].u[0] = *(const uint4*)(pl + r * 64 + j * 32 + hi * 8);             \
      pf[j].u[1] = *(const uint4*)(pl + r * 64 + j * 32 + 16 + hi * 8);        \
    }                                                                          \
    _Pragma("unroll") for (int n = 0; n < 4; ++n) {                            \
      _Pragma("unroll") for (int j = 0; j < 2; ++j) {                          \
        FragB vf;                                                              \
        const uint16_t* vrow = &(bufV)[(n * 16 + r) * 64 + j * 32 + hi * 8];   \
        vf.u[0] = *(const uint4*)(vrow);                                       \
        vf.u[1] = *(const uint4*)(vrow + 16);                                  \
        o[n] = __builtin_amdgcn_wmma_f32_16x16x32_bf16(                        \
            false, pf[j].v, false, vf.v, (short)0, o[n], false, false);        \
      }                                                                        \
    }                                                                          \
  }

  if (threadIdx.x < 32) FA_ISSUE(0, ldsK[0], ldsV[0])

  for (int key0 = 0; key0 < SEQ; key0 += 128) {
    if (threadIdx.x < 32) __builtin_amdgcn_s_wait_tensorcnt(0);
    __syncthreads();
    if (threadIdx.x < 32) FA_ISSUE(key0 + 64, ldsK[1], ldsV[1])  // always < SEQ
    FA_STEP(ldsK[0], ldsV[0])
    if (threadIdx.x < 32) __builtin_amdgcn_s_wait_tensorcnt(0);
    __syncthreads();
    if (threadIdx.x < 32 && (key0 + 128) < SEQ)
      FA_ISSUE(key0 + 128, ldsK[0], ldsV[0])
    FA_STEP(ldsK[1], ldsV[1])
  }

  // ---- normalize + store context (bf16, token-major [token, D]) ----
#pragma unroll
  for (int n = 0; n < 4; ++n)
#pragma unroll
    for (int i = 0; i < 8; ++i) {
      const float v = o[n][i] / l_i[i];
      Ctx[(size_t)(b * SEQ + q0 + i + hi * 8) * D_MODEL + h * HEAD_DIM +
          n * 16 + r] = f2bf(v);
    }
#undef FA_ISSUE
#undef FA_STEP
}

// ---------------------------------------------------------------------------
// Output projection: out = ctx @ Wo^T + bo  (fp32 out), TDM-staged tiles
// ---------------------------------------------------------------------------
__global__ __launch_bounds__(256) void gemm_out(const uint16_t* __restrict__ ctx,
                                                const uint16_t* __restrict__ wo,
                                                const float* __restrict__ bo,
                                                float* __restrict__ out) {
  __shared__ __align__(128) uint16_t ldsA[2][64 * 32];
  __shared__ __align__(128) uint16_t ldsB[2][256 * 32];

  const int wave = threadIdx.x >> 5;
  const int lane = threadIdx.x & 31;
  const int r = lane & 15;
  const int hi = lane >> 4;
  const int wm = wave >> 2;
  const int wn = wave & 3;
  const int m0 = blockIdx.x * 64 + wm * 32;
  const int n0 = blockIdx.y * 256 + wn * 64;

  const uint16_t* Asrc = ctx + (size_t)blockIdx.x * 64 * D_MODEL;
  const uint16_t* Wsrc = wo + (size_t)blockIdx.y * 256 * D_MODEL;

  v8f acc[2][4];
#pragma unroll
  for (int s = 0; s < 2; ++s)
#pragma unroll
    for (int n = 0; n < 4; ++n) acc[s][n] = (v8f){};

  GEMM_STAGED_BODY(Asrc, Wsrc, acc)

#pragma unroll
  for (int s = 0; s < 2; ++s)
#pragma unroll
    for (int n = 0; n < 4; ++n) {
      const int col = n0 + n * 16 + r;
      const float bb = bo[col];
#pragma unroll
      for (int i = 0; i < 8; ++i) {
        const int row = m0 + s * 16 + i + hi * 8;
        out[(size_t)row * D_MODEL + col] = acc[s][n][i] + bb;
      }
    }
}

// ---------------------------------------------------------------------------
// Host-side launcher
// ---------------------------------------------------------------------------
extern "C" void kernel_launch(void* const* d_in, const int* in_sizes, int n_in,
                              void* d_out, int out_size, void* d_ws,
                              size_t ws_size, hipStream_t stream) {
  (void)in_sizes; (void)n_in; (void)out_size; (void)ws_size;

  const float* x  = (const float*)d_in[0];
  const float* Wq = (const float*)d_in[1];
  const float* bq = (const float*)d_in[2];
  const float* Wk = (const float*)d_in[3];
  const float* bk = (const float*)d_in[4];
  const float* Wv = (const float*)d_in[5];
  const float* bv = (const float*)d_in[6];
  const float* Wo = (const float*)d_in[7];
  const float* bo = (const float*)d_in[8];
  float* out = (float*)d_out;

  const size_t XN = (size_t)ROWS * D_MODEL;
  const size_t WN = (size_t)D_MODEL * D_MODEL;

  uint16_t* xb  = (uint16_t*)d_ws;
  uint16_t* wqb = xb + XN;
  uint16_t* wkb = wqb + WN;
  uint16_t* wvb = wkb + WN;
  uint16_t* wob = wvb + WN;
  uint16_t* Qb  = wob + WN;
  uint16_t* Kb  = Qb + XN;
  uint16_t* Vt  = Kb + XN;   // per-head transposed V
  uint16_t* Ctx = Vt + XN;   // total ws usage: 88 MB

  cvt_f32_bf16<<<1024, 256, 0, stream>>>(x, xb, (int)XN);
  cvt_f32_bf16<<<512, 256, 0, stream>>>(Wq, wqb, (int)WN);
  cvt_f32_bf16<<<512, 256, 0, stream>>>(Wk, wkb, (int)WN);
  cvt_f32_bf16<<<512, 256, 0, stream>>>(Wv, wvb, (int)WN);
  cvt_f32_bf16<<<512, 256, 0, stream>>>(Wo, wob, (int)WN);

  dim3 gq(ROWS / 64, D_MODEL / 256, 3);
  gemm_qkv<<<gq, 256, 0, stream>>>(xb, wqb, wkb, wvb, bq, bk, bv, Qb, Kb, Vt);

  const int fblocks = (BATCH * NUM_HEADS * (SEQ / 16)) / 8;  // 1024
  flash_attn<<<fblocks, 256, 0, stream>>>(Qb, Kb, Vt, Ctx);

  dim3 go(ROWS / 64, D_MODEL / 256);
  gemm_out<<<go, 256, 0, stream>>>(Ctx, wob, bo, out);
}
